// EdgeUpdate_10222022164572
// MI455X (gfx1250) — matrix operand order repaired and checked by reading, not compiled, not run on hardware
//
#include <hip/hip_runtime.h>

typedef __attribute__((ext_vector_type(16))) __bf16 v16bf;
typedef __attribute__((ext_vector_type(8)))  __bf16 v8bf;
typedef __attribute__((ext_vector_type(4)))  __bf16 v4bf;
typedef __attribute__((ext_vector_type(8)))  float  v8f;

#define EDGE_DIM 32
#define NODE_DIM 64
#define IN_DIM   160   // 32 + 64 + 64
#define HID      128
#define OUT_DIM  32
#define TILE_E   128
#define XSTRIDE  168   // padded bf16 row stride for x-tile / h-stage
#define W1STRIDE 168   // Wt1 [n][k] stride (holds 160 K values)
#define W2STRIDE 136   // Wt2 [n][k] stride (holds 128 K values)

__device__ __forceinline__ v16bf cat16(v8bf lo, v8bf hi) {
    return __builtin_shufflevector(lo, hi, 0,1,2,3,4,5,6,7,8,9,10,11,12,13,14,15);
}

__global__ __launch_bounds__(256) void edge_mlp_kernel(
    const float* __restrict__ edges, const float* __restrict__ sender,
    const float* __restrict__ recv,  const float* __restrict__ W1,
    const float* __restrict__ b1,    const float* __restrict__ gamma,
    const float* __restrict__ beta,  const float* __restrict__ W2,
    const float* __restrict__ b2,    float* __restrict__ out, int E)
{
    __shared__ __bf16 xt [TILE_E  * XSTRIDE];   // 43008 B; x-tile, reused as h-stage
    __shared__ __bf16 wt1[HID     * W1STRIDE];  // 43008 B; W1^T [n][k] bf16
    __shared__ __bf16 wt2[OUT_DIM * W2STRIDE];  //  8704 B; W2^T [n][k] bf16

    const int tid  = threadIdx.x;
    const int wid  = tid >> 5;
    const int lane = tid & 31;
    const int l15  = lane & 15;
    const int half = lane >> 4;

    // ---- stage weights transposed to bf16 (once per block; coalesced global reads) ----
    for (int i = tid; i < IN_DIM * HID; i += 256) {
        int k = i / HID, n = i % HID;
        wt1[n * W1STRIDE + k] = (__bf16)W1[i];
    }
    for (int i = tid; i < HID * OUT_DIM; i += 256) {
        int k = i / OUT_DIM, n = i % OUT_DIM;
        wt2[n * W2STRIDE + k] = (__bf16)W2[i];
    }
    // per-lane params (indexed by output column n = j*16 + l15)
    float bb1[8], g[8], be[8];
    #pragma unroll
    for (int j = 0; j < 8; ++j) {
        bb1[j] = b1[j * 16 + l15];
        g[j]   = gamma[j * 16 + l15];
        be[j]  = beta[j * 16 + l15];
    }
    float bb2[2];
    bb2[0] = b2[l15];
    bb2[1] = b2[16 + l15];
    __syncthreads();

    const int ntiles = (E + TILE_E - 1) / TILE_E;
    for (int tile = blockIdx.x; tile < ntiles; tile += (int)gridDim.x) {
        const int base = tile * TILE_E;

        // ---- stage x-tile as bf16: [row][ edges(0:32) | sender(32:96) | recv(96:160) ] ----
        for (int i = tid; i < (TILE_E * EDGE_DIM) / 4; i += 256) {
            int row = (i * 4) / EDGE_DIM;
            int col = (i * 4) % EDGE_DIM;
            int rg  = base + row; if (rg >= E) rg = E - 1;
            const float4 v = *(const float4*)(edges + (size_t)rg * EDGE_DIM + col);
            v4bf p = { (__bf16)v.x, (__bf16)v.y, (__bf16)v.z, (__bf16)v.w };
            *(v4bf*)(xt + row * XSTRIDE + col) = p;
        }
        for (int i = tid; i < (TILE_E * NODE_DIM) / 4; i += 256) {
            int row = (i * 4) / NODE_DIM;
            int col = (i * 4) % NODE_DIM;
            int rg  = base + row; if (rg >= E) rg = E - 1;
            float4 v = *(const float4*)(sender + (size_t)rg * NODE_DIM + col);
            v4bf p = { (__bf16)v.x, (__bf16)v.y, (__bf16)v.z, (__bf16)v.w };
            *(v4bf*)(xt + row * XSTRIDE + 32 + col) = p;
            v = *(const float4*)(recv + (size_t)rg * NODE_DIM + col);
            v4bf q = { (__bf16)v.x, (__bf16)v.y, (__bf16)v.z, (__bf16)v.w };
            *(v4bf*)(xt + row * XSTRIDE + 96 + col) = q;
        }
        __syncthreads();

        // ---- layer 1: each wave owns 16 rows; A frags per ISA layout ----
        const int mrow = wid * 16 + l15;
        v16bf afrag[5];
        #pragma unroll
        for (int t = 0; t < 5; ++t) {
            const __bf16* p = xt + mrow * XSTRIDE + t * 32 + half * 8;
            v8bf lo = *(const v8bf*)p;        // K = t*32 + half*8 + 0..7
            v8bf hi = *(const v8bf*)(p + 16); // K = t*32 + 16 + half*8 + 0..7
            afrag[t] = cat16(lo, hi);
        }

        v8f acc[8];
        #pragma unroll
        for (int j = 0; j < 8; ++j) {
            v8f c;
            #pragma unroll
            for (int r = 0; r < 8; ++r) c[r] = bb1[j];
            #pragma unroll
            for (int t = 0; t < 5; ++t) {
                const __bf16* p = wt1 + (j * 16 + l15) * W1STRIDE + t * 32 + half * 8;
                v8bf lo = *(const v8bf*)p;       // K = t*32 + half*8 + 0..7 of column n
                v8bf hi = *(const v8bf*)(p + 16); // K = t*32 + 16 + half*8 + 0..7
                v16bf bfrag = cat16(lo, hi);
                c = __builtin_amdgcn_wmma_f32_16x16x32_bf16(
                        false, afrag[t], false, bfrag, (short)0, c, false, false);
            }
            acc[j] = c;
        }

        // ---- LayerNorm over HID=128 per row (rows split by lane half) ----
        float sum[8], sq[8];
        #pragma unroll
        for (int r = 0; r < 8; ++r) {
            float s = 0.f, s2 = 0.f;
            #pragma unroll
            for (int j = 0; j < 8; ++j) { float v = acc[j][r]; s += v; s2 += v * v; }
            #pragma unroll
            for (int m = 8; m >= 1; m >>= 1) {
                s  += __shfl_xor(s,  m, 16);
                s2 += __shfl_xor(s2, m, 16);
            }
            sum[r] = s; sq[r] = s2;
        }
        const float inv128 = 1.0f / 128.0f;
        float mu[8], rs[8];
        #pragma unroll
        for (int r = 0; r < 8; ++r) {
            mu[r] = sum[r] * inv128;
            float var = sq[r] * inv128 - mu[r] * mu[r];
            rs[r] = rsqrtf(var + 1e-5f);
        }

        // ---- normalize + relu; stage h (bf16) into wave-private slice of xt ----
        #pragma unroll
        for (int j = 0; j < 8; ++j) {
            #pragma unroll
            for (int r = 0; r < 8; ++r) {
                float v = (acc[j][r] - mu[r]) * rs[r] * g[j] + be[j];
                v = fmaxf(v, 0.0f);
                int hrow = wid * 16 + half * 8 + r;   // D layout: halves hold rows 0-7 / 8-15
                xt[hrow * XSTRIDE + j * 16 + l15] = (__bf16)v;
            }
        }

        // ---- layer 2: out[16][32] = h[16][128] @ W2 ----
        v16bf afrag2[4];
        #pragma unroll
        for (int t = 0; t < 4; ++t) {
            const __bf16* p = xt + mrow * XSTRIDE + t * 32 + half * 8;
            v8bf lo = *(const v8bf*)p;
            v8bf hi = *(const v8bf*)(p + 16);
            afrag2[t] = cat16(lo, hi);
        }
        #pragma unroll
        for (int jj = 0; jj < 2; ++jj) {
            v8f c;
            #pragma unroll
            for (int r = 0; r < 8; ++r) c[r] = bb2[jj];
            #pragma unroll
            for (int t = 0; t < 4; ++t) {
                const __bf16* p = wt2 + (jj * 16 + l15) * W2STRIDE + t * 32 + half * 8;
                v8bf lo = *(const v8bf*)p;
                v8bf hi = *(const v8bf*)(p + 16);
                v16bf bfrag = cat16(lo, hi);
                c = __builtin_amdgcn_wmma_f32_16x16x32_bf16(
                        false, afrag2[t], false, bfrag, (short)0, c, false, false);
            }
            // coalesced stores: lanes of a half store 16 contiguous f32 of one row
            #pragma unroll
            for (int r = 0; r < 8; ++r) {
                int row = base + wid * 16 + half * 8 + r;
                if (row < E) out[(size_t)row * OUT_DIM + jj * 16 + l15] = c[r];
            }
        }
        __syncthreads();   // protect xt (x-tile + h-stage) before next iteration
    }
}

extern "C" void kernel_launch(void* const* d_in, const int* in_sizes, int n_in,
                              void* d_out, int out_size, void* d_ws, size_t ws_size,
                              hipStream_t stream) {
    const float* edges  = (const float*)d_in[0];
    const float* sender = (const float*)d_in[1];
    const float* recv   = (const float*)d_in[2];
    const float* W1     = (const float*)d_in[3];
    const float* b1     = (const float*)d_in[4];
    const float* gamma  = (const float*)d_in[5];
    const float* beta   = (const float*)d_in[6];
    const float* W2     = (const float*)d_in[7];
    const float* b2     = (const float*)d_in[8];
    float* out = (float*)d_out;

    const int E = in_sizes[0] / EDGE_DIM;
    const int ntiles = (E + TILE_E - 1) / TILE_E;
    const int grid = ntiles < 1024 ? ntiles : 1024;
    edge_mlp_kernel<<<grid, 256, 0, stream>>>(edges, sender, recv, W1, b1,
                                              gamma, beta, W2, b2, out, E);
}
